// TransformerBlock_52750788329631
// MI455X (gfx1250) — compile-verified, matrix operand order checked by reading, not run on hardware
//
#include <hip/hip_runtime.h>
#include <stdint.h>

// ---- problem constants (match reference) ----
#define DIMN 1024
#define NHEAD 16
#define HDIM 64
#define FFND 2816
#define GRP 128
#define NB 2
#define TT 2048
#define BT (NB * TT) // 4096

typedef unsigned short u16;
typedef unsigned int u32;

typedef __attribute__((ext_vector_type(16))) __bf16 v16bf;
typedef __attribute__((ext_vector_type(8))) float v8f;
typedef __attribute__((ext_vector_type(4))) unsigned int uif4;

// ---- bf16 helpers (round-to-nearest-even), stored as u16 ----
__device__ __forceinline__ u16 f2bf(float f) {
    u32 u = __float_as_uint(f);
    u32 r = u + 0x7fffu + ((u >> 16) & 1u);
    return (u16)(r >> 16);
}
__device__ __forceinline__ float bf2f(u16 h) { return __uint_as_float(((u32)h) << 16); }

// ---- WMMA fragment loader (ISA 7.12.2 wave32 16-bit 16x32 layout) ----
// Tile is row-major [16 rows x 32 k], row r = lane%16, half = lane/16.
// Frag elems 0..7  = k = half*8 + 0..7        (one b128 load)
// Frag elems 8..15 = k = 16 + half*8 + 0..7   (one b128 load)
__device__ __forceinline__ v16bf load_frag(const u16* p, int ld) {
    int lane = threadIdx.x & 31;
    int r = lane & 15;
    int hf = lane >> 4;
    const u16* row = p + (size_t)r * ld + hf * 8;
    union { uif4 q[2]; v16bf v; } u;
    u.q[0] = *(const uif4*)(row);
    u.q[1] = *(const uif4*)(row + 16);
    return u.v;
}

__device__ __forceinline__ v8f wmma_bf16(v16bf a, v16bf b, v8f c) {
    return __builtin_amdgcn_wmma_f32_16x16x32_bf16(false, a, false, b, (short)0, c, false, false);
}

// ---- BitLinear group-wise ternary dequant -> bf16 (one wave per (o,group)) ----
__global__ void __launch_bounds__(256) dequant_kernel(const float* __restrict__ w,
                                                      u16* __restrict__ out, int O, int I) {
    int warp = (blockIdx.x * 256 + threadIdx.x) >> 5;
    int lane = threadIdx.x & 31;
    int ng = I / GRP;
    int o = warp / ng;
    int g = warp - o * ng;
    const float* p = w + (size_t)o * I + g * GRP;
    float v0 = p[lane], v1 = p[lane + 32], v2 = p[lane + 64], v3 = p[lane + 96];
    float s = fabsf(v0) + fabsf(v1) + fabsf(v2) + fabsf(v3);
#pragma unroll
    for (int m = 16; m >= 1; m >>= 1) s += __shfl_xor(s, m, 32);
    float scale = s * (1.0f / (float)GRP);
    float inv = 1.0f / (scale + 1e-5f);
    u16* q = out + (size_t)o * I + g * GRP;
    q[lane]      = f2bf(fminf(1.f, fmaxf(-1.f, rintf(v0 * inv))) * scale);
    q[lane + 32] = f2bf(fminf(1.f, fmaxf(-1.f, rintf(v1 * inv))) * scale);
    q[lane + 64] = f2bf(fminf(1.f, fmaxf(-1.f, rintf(v2 * inv))) * scale);
    q[lane + 96] = f2bf(fminf(1.f, fmaxf(-1.f, rintf(v3 * inv))) * scale);
}

// ---- mhc_prepare + RMSNorm fused, fp32 -> bf16 (one block per token) ----
__global__ void __launch_bounds__(256) prep_rms_kernel(const float* __restrict__ x,
                                                       const float* __restrict__ hin,
                                                       const float* __restrict__ nw,
                                                       u16* __restrict__ out) {
    int tok = blockIdx.x;
    const float* x0 = x + (size_t)tok * 2 * DIMN;
    const float* x1 = x0 + DIMN;
    float h0 = hin[0], h1 = hin[1];
    float vc[4];
    float ss = 0.f;
#pragma unroll
    for (int i = 0; i < 4; i++) {
        int d = threadIdx.x + i * 256;
        float v = h0 * x0[d] + h1 * x1[d];
        vc[i] = v;
        ss += v * v;
    }
#pragma unroll
    for (int m = 16; m >= 1; m >>= 1) ss += __shfl_xor(ss, m, 32);
    __shared__ float red[8];
    if ((threadIdx.x & 31) == 0) red[threadIdx.x >> 5] = ss;
    __syncthreads();
    float tot = red[0] + red[1] + red[2] + red[3] + red[4] + red[5] + red[6] + red[7];
    float norm = rsqrtf(tot * (1.0f / (float)DIMN) + 1e-6f);
#pragma unroll
    for (int i = 0; i < 4; i++) {
        int d = threadIdx.x + i * 256;
        out[(size_t)tok * DIMN + d] = f2bf(vc[i] * norm * nw[d]);
    }
}

// ---- generic WMMA GEMM: C[M,N] = A[M,K](bf16) x W[N,K](bf16)^T ----
// Wave tile: 32 (M) x 64 (N): 2 A-frags x 4 B-frags -> 8 WMMAs per 32-wide k-step.
// mode 0: store fp32 Cf; mode 1: store bf16 Ch; mode 2: store bf16 transposed (V^T layout)
__global__ void __launch_bounds__(128) gemm_bf16(const u16* __restrict__ A,
                                                 const u16* __restrict__ W,
                                                 float* __restrict__ Cf, u16* __restrict__ Ch,
                                                 int M, int N, int K, int mode) {
    int wave = threadIdx.x >> 5;
    int lane = threadIdx.x & 31;
    int n0 = (blockIdx.x * 4 + wave) * 64;
    int m0 = blockIdx.y * 32;
    const u16* Ab0 = A + (size_t)m0 * K;
    const u16* Ab1 = A + (size_t)(m0 + 16) * K;
    const u16* Wb = W + (size_t)n0 * K;
    v8f c0[4], c1[4];
#pragma unroll
    for (int t = 0; t < 4; ++t) { c0[t] = (v8f){}; c1[t] = (v8f){}; }
    for (int k0 = 0; k0 < K; k0 += 32) {
        __builtin_prefetch(Ab0 + k0 + 256, 0, 0); // global_prefetch_b8
        __builtin_prefetch(Ab1 + k0 + 256, 0, 0);
        v16bf a0 = load_frag(Ab0 + k0, K);
        v16bf a1 = load_frag(Ab1 + k0, K);
#pragma unroll
        for (int t = 0; t < 4; ++t) {
            v16bf b = load_frag(Wb + (size_t)(t * 16) * K + k0, K);
            c0[t] = wmma_bf16(a0, b, c0[t]);
            c1[t] = wmma_bf16(a1, b, c1[t]);
        }
    }
    int col = lane & 15, hf = lane >> 4;
#pragma unroll
    for (int half = 0; half < 2; ++half) {
#pragma unroll
        for (int v = 0; v < 8; ++v) {
            int row = m0 + half * 16 + hf * 8 + v;
            float e0 = half ? c1[0][v] : c0[0][v];
            float e1 = half ? c1[1][v] : c0[1][v];
            float e2 = half ? c1[2][v] : c0[2][v];
            float e3 = half ? c1[3][v] : c0[3][v];
            if (mode == 0) {
                float* o = Cf + (size_t)row * N + n0 + col;
                o[0] = e0; o[16] = e1; o[32] = e2; o[48] = e3;
            } else if (mode == 1) {
                u16* o = Ch + (size_t)row * N + n0 + col;
                o[0] = f2bf(e0); o[16] = f2bf(e1); o[32] = f2bf(e2); o[48] = f2bf(e3);
            } else {
                // transposed store: vt[(b*DIMN + n)*TT + t], row = b*TT + t
                int bb = row >> 11, t = row & (TT - 1);
                size_t base = (size_t)bb * DIMN * TT;
                Ch[base + (size_t)(n0 + col) * TT + t]      = f2bf(e0);
                Ch[base + (size_t)(n0 + col + 16) * TT + t] = f2bf(e1);
                Ch[base + (size_t)(n0 + col + 32) * TT + t] = f2bf(e2);
                Ch[base + (size_t)(n0 + col + 48) * TT + t] = f2bf(e3);
            }
        }
    }
}

// ---- RoPE on fp32 Q/K, emits bf16 ----
__global__ void __launch_bounds__(256) rope_kernel(const float* __restrict__ qf,
                                                   const float* __restrict__ kf,
                                                   const float* __restrict__ fc,
                                                   const float* __restrict__ fs,
                                                   u16* __restrict__ qh, u16* __restrict__ kh) {
    int idx = blockIdx.x * 256 + threadIdx.x; // BT*16*32 threads
    int pr = idx & 31;
    int h = (idx >> 5) & 15;
    int tok = idx >> 9;
    int t = tok & (TT - 1);
    float c = fc[t * 32 + pr], s = fs[t * 32 + pr];
    size_t base = (size_t)tok * DIMN + h * HDIM + pr * 2;
    float qr = qf[base], qi = qf[base + 1];
    qh[base]     = f2bf(qr * c - qi * s);
    qh[base + 1] = f2bf(qr * s + qi * c);
    float kr = kf[base], ki = kf[base + 1];
    kh[base]     = f2bf(kr * c - ki * s);
    kh[base + 1] = f2bf(kr * s + ki * c);
}

// ---- flash attention: one wave per (b, h, 16-row q tile); causal, online softmax ----
__global__ void __launch_bounds__(128) flash_kernel(const u16* __restrict__ qh,
                                                    const u16* __restrict__ kh,
                                                    const u16* __restrict__ vt,
                                                    u16* __restrict__ ctx) {
    __shared__ __align__(16) u16 ptile[4][16 * 32];
    int wave = threadIdx.x >> 5, lane = threadIdx.x & 31;
    int g = blockIdx.x * 4 + wave;
    int qt = g & 127;
    int bh = g >> 7;
    int b = bh >> 4, h = bh & 15;
    int t0 = qt * 16;
    int col = lane & 15, hf = lane >> 4;

    const u16* qp = qh + ((size_t)(b * TT + t0)) * DIMN + h * HDIM;
    v16bf aQ0 = load_frag(qp, DIMN);
    v16bf aQ1 = load_frag(qp + 32, DIMN);

    v8f o0 = {}, o1 = {}, o2 = {}, o3 = {};
    float mrow[8], lrow[8];
#pragma unroll
    for (int v = 0; v < 8; v++) { mrow[v] = -1e30f; lrow[v] = 0.f; }

    int nch = (t0 >> 5) + 1; // kv chunks of 32 needed (causal)
    u16* pl = &ptile[wave][0];
    const float sc = 0.125f; // 1/sqrt(64)

    for (int c = 0; c < nch; ++c) {
        int kv0 = c * 32;
        const u16* kp = kh + ((size_t)(b * TT + kv0)) * DIMN + h * HDIM;
        v8f z0 = {}, z1 = {};
        v8f s0 = wmma_bf16(aQ0, load_frag(kp, DIMN), z0);
        s0 = wmma_bf16(aQ1, load_frag(kp + 32, DIMN), s0);
        v8f s1 = wmma_bf16(aQ0, load_frag(kp + (size_t)16 * DIMN, DIMN), z1);
        s1 = wmma_bf16(aQ1, load_frag(kp + (size_t)16 * DIMN + 32, DIMN), s1);

#pragma unroll
        for (int v = 0; v < 8; v++) {
            int qpos = t0 + hf * 8 + v;
            float a0 = (kv0 + col > qpos) ? -1e30f : s0[v] * sc;
            float a1 = (kv0 + 16 + col > qpos) ? -1e30f : s1[v] * sc;
            float mx = fmaxf(a0, a1);
#pragma unroll
            for (int m = 8; m >= 1; m >>= 1) mx = fmaxf(mx, __shfl_xor(mx, m, 32));
            float mn = fmaxf(mrow[v], mx);
            float al = __expf(mrow[v] - mn);
            mrow[v] = mn;
            float p0 = __expf(a0 - mn), p1 = __expf(a1 - mn);
            float rs = p0 + p1;
#pragma unroll
            for (int m = 8; m >= 1; m >>= 1) rs += __shfl_xor(rs, m, 32);
            lrow[v] = lrow[v] * al + rs;
            o0[v] *= al; o1[v] *= al; o2[v] *= al; o3[v] *= al;
            pl[(hf * 8 + v) * 32 + col]      = f2bf(p0);
            pl[(hf * 8 + v) * 32 + 16 + col] = f2bf(p1);
        }
        __builtin_amdgcn_fence(__ATOMIC_ACQ_REL, "wavefront");
        v16bf aP = load_frag(pl, 32); // P re-layout via LDS (ds_load_b128)
        const u16* vp = vt + ((size_t)(b * DIMN + h * HDIM)) * TT + kv0;
        o0 = wmma_bf16(aP, load_frag(vp, TT), o0);
        o1 = wmma_bf16(aP, load_frag(vp + (size_t)16 * TT, TT), o1);
        o2 = wmma_bf16(aP, load_frag(vp + (size_t)32 * TT, TT), o2);
        o3 = wmma_bf16(aP, load_frag(vp + (size_t)48 * TT, TT), o3);
    }

    u16* op = ctx + ((size_t)(b * TT + t0)) * DIMN + h * HDIM;
#pragma unroll
    for (int v = 0; v < 8; v++) {
        float inv = 1.0f / lrow[v];
        size_t r = (size_t)(hf * 8 + v) * DIMN;
        op[r + col]      = f2bf(o0[v] * inv);
        op[r + 16 + col] = f2bf(o1[v] * inv);
        op[r + 32 + col] = f2bf(o2[v] * inv);
        op[r + 48 + col] = f2bf(o3[v] * inv);
    }
}

// ---- SwiGLU gate: h = silu(g) * u (bf16 in/out) ----
__global__ void __launch_bounds__(256) silu_mul_kernel(const u16* __restrict__ g,
                                                       const u16* __restrict__ u,
                                                       u16* __restrict__ o) {
    int i = blockIdx.x * 256 + threadIdx.x;
    float gv = bf2f(g[i]);
    float uv = bf2f(u[i]);
    o[i] = f2bf(gv / (1.0f + __expf(-gv)) * uv);
}

// ---- mhc_apply: out[b,t,r,d] = sum_s hres[r,s]*x[b,t,s,d] + hout[r]*y[b,t,d] ----
__global__ void __launch_bounds__(256) mhc_apply_kernel(const float* __restrict__ x,
                                                        const float* __restrict__ y,
                                                        const float* __restrict__ hres,
                                                        const float* __restrict__ hout,
                                                        float* __restrict__ out) {
    size_t i = (size_t)blockIdx.x * 256 + threadIdx.x; // BT*2*DIMN
    int d = (int)(i & (DIMN - 1));
    int r = (int)((i >> 10) & 1);
    size_t tok = i >> 11;
    float v = hres[r * 2 + 0] * x[tok * 2 * DIMN + d] +
              hres[r * 2 + 1] * x[tok * 2 * DIMN + DIMN + d] +
              hout[r] * y[tok * DIMN + d];
    out[i] = v;
}

extern "C" void kernel_launch(void* const* d_in, const int* in_sizes, int n_in,
                              void* d_out, int out_size, void* d_ws, size_t ws_size,
                              hipStream_t stream) {
    (void)in_sizes; (void)n_in; (void)out_size; (void)ws_size;
    const float* x_exp  = (const float*)d_in[0];
    const float* fcos   = (const float*)d_in[1];
    const float* fsin   = (const float*)d_in[2];
    const float* h_in_a = (const float*)d_in[3];
    const float* h_res_a= (const float*)d_in[4];
    const float* h_out_a= (const float*)d_in[5];
    const float* h_in_f = (const float*)d_in[6];
    const float* h_res_f= (const float*)d_in[7];
    const float* h_out_f= (const float*)d_in[8];
    const float* nw_a   = (const float*)d_in[9];
    const float* nw_f   = (const float*)d_in[10];
    const float* wq = (const float*)d_in[11];
    const float* wk = (const float*)d_in[12];
    const float* wv = (const float*)d_in[13];
    const float* wo = (const float*)d_in[14];
    const float* wg = (const float*)d_in[15];
    const float* wu = (const float*)d_in[16];
    const float* wd = (const float*)d_in[17];

    char* ws = (char*)d_ws;
    size_t off = 0;
    auto alloc = [&](size_t bytes) -> char* {
        char* p = ws + off;
        off = (off + bytes + 255) & ~(size_t)255;
        return p;
    };
    u16* wqh = (u16*)alloc((size_t)DIMN * DIMN * 2);
    u16* wkh = (u16*)alloc((size_t)DIMN * DIMN * 2);
    u16* wvh = (u16*)alloc((size_t)DIMN * DIMN * 2);
    u16* woh = (u16*)alloc((size_t)DIMN * DIMN * 2);
    u16* wgh = (u16*)alloc((size_t)FFND * DIMN * 2);
    u16* wuh = (u16*)alloc((size_t)FFND * DIMN * 2);
    u16* wdh = (u16*)alloc((size_t)DIMN * FFND * 2);
    u16* xn  = (u16*)alloc((size_t)BT * DIMN * 2);
    float* qf = (float*)alloc((size_t)BT * DIMN * 4);
    float* kf = (float*)alloc((size_t)BT * DIMN * 4);
    u16* qhb = (u16*)alloc((size_t)BT * DIMN * 2);
    u16* khb = (u16*)alloc((size_t)BT * DIMN * 2);
    u16* vtb = (u16*)alloc((size_t)BT * DIMN * 2); // V transposed [b, d, t]
    u16* ctx = (u16*)alloc((size_t)BT * DIMN * 2);
    float* yb = (float*)alloc((size_t)BT * DIMN * 4);
    float* x2 = (float*)alloc((size_t)BT * 2 * DIMN * 4);
    u16* gh = (u16*)alloc((size_t)BT * FFND * 2);
    u16* uh = (u16*)alloc((size_t)BT * FFND * 2);
    u16* hh = (u16*)alloc((size_t)BT * FFND * 2);

    // 1) quantize weights (ternary group-128) -> bf16
    dequant_kernel<<<1024, 256, 0, stream>>>(wq, wqh, DIMN, DIMN);
    dequant_kernel<<<1024, 256, 0, stream>>>(wk, wkh, DIMN, DIMN);
    dequant_kernel<<<1024, 256, 0, stream>>>(wv, wvh, DIMN, DIMN);
    dequant_kernel<<<1024, 256, 0, stream>>>(wo, woh, DIMN, DIMN);
    dequant_kernel<<<2816, 256, 0, stream>>>(wg, wgh, FFND, DIMN);
    dequant_kernel<<<2816, 256, 0, stream>>>(wu, wuh, FFND, DIMN);
    dequant_kernel<<<2816, 256, 0, stream>>>(wd, wdh, DIMN, FFND);

    // 2) attn input: mhc_prepare + rmsnorm
    prep_rms_kernel<<<BT, 256, 0, stream>>>(x_exp, h_in_a, nw_a, xn);

    // 3) QKV projections (WMMA), 32x64 wave tiles
    dim3 g1(DIMN / 256, BT / 32); // (4, 128)
    gemm_bf16<<<g1, 128, 0, stream>>>(xn, wqh, qf, nullptr, BT, DIMN, DIMN, 0);
    gemm_bf16<<<g1, 128, 0, stream>>>(xn, wkh, kf, nullptr, BT, DIMN, DIMN, 0);
    gemm_bf16<<<g1, 128, 0, stream>>>(xn, wvh, nullptr, vtb, BT, DIMN, DIMN, 2);

    // 4) RoPE -> bf16
    rope_kernel<<<(BT * NHEAD * 32) / 256, 256, 0, stream>>>(qf, kf, fcos, fsin, qhb, khb);

    // 5) causal flash attention (WMMA QK^T and PV)
    flash_kernel<<<(NB * NHEAD * (TT / 16)) / 4, 128, 0, stream>>>(qhb, khb, vtb, ctx);

    // 6) output projection
    gemm_bf16<<<g1, 128, 0, stream>>>(ctx, woh, yb, nullptr, BT, DIMN, DIMN, 0);

    // 7) mhc_apply (attn)
    mhc_apply_kernel<<<(BT * 2 * DIMN) / 256, 256, 0, stream>>>(x_exp, yb, h_res_a, h_out_a, x2);

    // 8) ffn input: mhc_prepare + rmsnorm
    prep_rms_kernel<<<BT, 256, 0, stream>>>(x2, h_in_f, nw_f, xn);

    // 9) gate/up projections (WMMA), SwiGLU, down projection (WMMA)
    dim3 g2(FFND / 256, BT / 32); // (11, 128)
    gemm_bf16<<<g2, 128, 0, stream>>>(xn, wgh, nullptr, gh, BT, FFND, DIMN, 1);
    gemm_bf16<<<g2, 128, 0, stream>>>(xn, wuh, nullptr, uh, BT, FFND, DIMN, 1);
    silu_mul_kernel<<<(BT * FFND) / 256, 256, 0, stream>>>(gh, uh, hh);
    gemm_bf16<<<g1, 128, 0, stream>>>(hh, wdh, yb, nullptr, BT, DIMN, FFND, 0);

    // 10) mhc_apply (ffn) -> final output
    mhc_apply_kernel<<<(BT * 2 * DIMN) / 256, 256, 0, stream>>>(x2, yb, h_res_f, h_out_f, (float*)d_out);
}
